// PLoss_my2_83133386981799
// MI455X (gfx1250) — matrix operand order repaired and checked by reading, not compiled, not run on hardware
//
#include <hip/hip_runtime.h>
#include <hip/hip_bf16.h>
#include <math.h>

// ---- problem constants (match reference) ----
#define KDIM 1024
#define BTOT 131072
#define NPOS 65536
#define NUNL 65536
#define NSEL 6553            // int(NU * 0.1)
#define NTOT (NPOS + NSEL)   // 72089
#define ASTR 1032            // padded LDS row stride (halves): conflict-free ds_load_b128

typedef __attribute__((ext_vector_type(16))) _Float16 v16h;
typedef __attribute__((ext_vector_type(8)))  _Float16 v8h;
typedef __attribute__((ext_vector_type(4)))  _Float16 v4h;
typedef __attribute__((ext_vector_type(8)))  float    v8f;

// ------------------------------------------------------------------
// Kernel 1: global_logit f32 -> f16, plus g2[c] = ||g_c||^2
// ------------------------------------------------------------------
__global__ void k_prep_gl(const float* __restrict__ g,
                          _Float16* __restrict__ gl16,
                          float* __restrict__ g2) {
  __shared__ float red[256];
  const int row = blockIdx.x;
  const int t = threadIdx.x;
  const float4 v = ((const float4*)(g + (size_t)row * KDIM))[t];
  v4h h;
  h.x = (_Float16)v.x; h.y = (_Float16)v.y; h.z = (_Float16)v.z; h.w = (_Float16)v.w;
  *(v4h*)(gl16 + (size_t)row * KDIM + t * 4) = h;
  red[t] = v.x * v.x + v.y * v.y + v.z * v.z + v.w * v.w;
  __syncthreads();
  for (int s = 128; s > 0; s >>= 1) {
    if (t < s) red[t] += red[t + s];
    __syncthreads();
  }
  if (t == 0) g2[row] = red[0];
}

// ------------------------------------------------------------------
// Kernel 2: WMMA distance min/argmin.
// Block: 256 thr (8 waves), 64 U-rows. Each wave owns TWO 16-col
// strips of G per pass (8 accumulators) so every ds_load_b128
// A-fragment feeds 2 WMMAs: 32 FLOP per LDS byte.
// ------------------------------------------------------------------
__global__ void __launch_bounds__(256) k_dist(
    const float* __restrict__ outputs, const _Float16* __restrict__ gl16,
    const float* __restrict__ g2, float* __restrict__ mind, int* __restrict__ mini) {
  __shared__ _Float16 As[64 * ASTR];    // 132096 B  f16 A panel
  __shared__ float ssq[64][4];
  __shared__ float u2row[64];
  __shared__ float wmv[64 * 8];
  __shared__ int   wmi[64 * 8];

  const int t    = threadIdx.x;
  const int lane = t & 31;
  const int wave = t >> 5;
  const int blk  = blockIdx.x;
  const size_t urow0 = (size_t)NPOS + (size_t)blk * 64;

  // ---- stage A panel: f32 -> f16 into LDS, accumulate row sum-of-squares ----
  {
    const int r = t >> 2;   // 0..63
    const int s = t & 3;    // 0..3
    const float4* src = (const float4*)(outputs + (urow0 + r) * KDIM);
    float acc = 0.f;
    for (int i = 0; i < 64; ++i) {
      const int c4 = s + (i << 2);            // float4 index 0..255
      const float4 v = src[c4];
      acc += v.x * v.x + v.y * v.y + v.z * v.z + v.w * v.w;
      v4h h;
      h.x = (_Float16)v.x; h.y = (_Float16)v.y; h.z = (_Float16)v.z; h.w = (_Float16)v.w;
      *(v4h*)(&As[r * ASTR + (c4 << 2)]) = h;
    }
    ssq[r][s] = acc;
  }
  __syncthreads();
  if (t < 64) u2row[t] = ssq[t][0] + ssq[t][1] + ssq[t][2] + ssq[t][3];

  float rmin[4][8];
  int   ridx[4][8];
#pragma unroll
  for (int a = 0; a < 4; ++a)
#pragma unroll
    for (int b = 0; b < 8; ++b) { rmin[a][b] = 3.4e38f; ridx[a][b] = 0; }

  const int lmod  = lane & 15;
  const int hi    = lane >> 4;          // 0|1
  const int kadjB = hi ? 16 : 0;        // B: high lanes hold K=16..31
  const int kadjA = hi ? 16 : 0;        // A: high lanes hold K=16..31

#define LOADA(T)                                                          \
  { const _Float16* ap_ = &As[((T)*16 + lmod) * ASTR + k0 + kadjA];       \
    ((v8h*)&af)[0] = *(const v8h*)(ap_);                                  \
    ((v8h*)&af)[1] = *(const v8h*)(ap_ + 8); }

#define MERGE(ACC, T, BCOL, G2V)                                          \
  { _Pragma("unroll")                                                     \
    for (int j = 0; j < 8; ++j) {                                         \
      float v_ = __builtin_fmaf(-2.0f, (ACC)[j], (G2V));                  \
      if (v_ < rmin[T][j]) { rmin[T][j] = v_; ridx[T][j] = (BCOL); }      \
    } }

  for (int it = 0; it < 4; ++it) {
    const int s0 = wave + (it << 4);               // strip pair: s0, s0+8
    const int bcol0 = (s0 << 4) + lmod;            // this lane's G columns
    const int bcol1 = bcol0 + 128;                 // (s0+8)*16 + lmod
    const _Float16* bb0 = gl16 + (size_t)bcol0 * KDIM + kadjB;
    const _Float16* bb1 = gl16 + (size_t)bcol1 * KDIM + kadjB;
    v8f a00 = {}, a01 = {}, a02 = {}, a03 = {};
    v8f a10 = {}, a11 = {}, a12 = {}, a13 = {};
    for (int k0 = 0; k0 < KDIM; k0 += 32) {
      const v16h bf0 = *(const v16h*)(bb0 + k0);   // 2x global_load_b128, L2-hot
      const v16h bf1 = *(const v16h*)(bb1 + k0);
      v16h af;
      LOADA(0);
      a00 = __builtin_amdgcn_wmma_f32_16x16x32_f16(false, af, false, bf0, (short)0, a00, false, false);
      a10 = __builtin_amdgcn_wmma_f32_16x16x32_f16(false, af, false, bf1, (short)0, a10, false, false);
      LOADA(1);
      a01 = __builtin_amdgcn_wmma_f32_16x16x32_f16(false, af, false, bf0, (short)0, a01, false, false);
      a11 = __builtin_amdgcn_wmma_f32_16x16x32_f16(false, af, false, bf1, (short)0, a11, false, false);
      LOADA(2);
      a02 = __builtin_amdgcn_wmma_f32_16x16x32_f16(false, af, false, bf0, (short)0, a02, false, false);
      a12 = __builtin_amdgcn_wmma_f32_16x16x32_f16(false, af, false, bf1, (short)0, a12, false, false);
      LOADA(3);
      a03 = __builtin_amdgcn_wmma_f32_16x16x32_f16(false, af, false, bf0, (short)0, a03, false, false);
      a13 = __builtin_amdgcn_wmma_f32_16x16x32_f16(false, af, false, bf1, (short)0, a13, false, false);
    }
    const float g2v0 = g2[bcol0];
    const float g2v1 = g2[bcol1];
    MERGE(a00, 0, bcol0, g2v0); MERGE(a01, 1, bcol0, g2v0);
    MERGE(a02, 2, bcol0, g2v0); MERGE(a03, 3, bcol0, g2v0);
    MERGE(a10, 0, bcol1, g2v1); MERGE(a11, 1, bcol1, g2v1);
    MERGE(a12, 2, bcol1, g2v1); MERGE(a13, 3, bcol1, g2v1);
  }
#undef LOADA
#undef MERGE

  // ---- cross-lane min/argmin within each 16-lane column group ----
#pragma unroll
  for (int tt = 0; tt < 4; ++tt) {
#pragma unroll
    for (int j = 0; j < 8; ++j) {
      float v = rmin[tt][j];
      int   ix = ridx[tt][j];
#pragma unroll
      for (int off = 8; off >= 1; off >>= 1) {
        const float ov = __shfl_xor(v, off, 32);
        const int   oi = __shfl_xor(ix, off, 32);
        if (ov < v || (ov == v && oi < ix)) { v = ov; ix = oi; }
      }
      if (lmod == 0) {
        const int row = tt * 16 + j + (hi ? 8 : 0);
        wmv[row * 8 + wave] = v;
        wmi[row * 8 + wave] = ix;
      }
    }
  }
  __syncthreads();

  // ---- cross-wave merge + finalize distance ----
  if (t < 64) {
    float best = wmv[t * 8];
    int   bix  = wmi[t * 8];
    for (int w = 1; w < 8; ++w) {
      const float v = wmv[t * 8 + w];
      const int   ix = wmi[t * 8 + w];
      if (v < best || (v == best && ix < bix)) { best = v; bix = ix; }
    }
    const float d2 = u2row[t] + best;
    const int u = blk * 64 + t;
    mind[u] = sqrtf(fmaxf(d2, 0.f));
    mini[u] = bix;
  }
}

// ------------------------------------------------------------------
// Top-k smallest (two-level radix select on monotone uint view)
// ------------------------------------------------------------------
__global__ void k_zero_hist(int* hist, int* prm) {
  const int t = threadIdx.x;
  if (t < 1024) hist[t] = 0;
  if (t < 16) prm[t] = 0;
}

__global__ void k_hist1(const float* __restrict__ mind, int* __restrict__ hist) {
  const int u = blockIdx.x * 256 + threadIdx.x;
  const unsigned b = __float_as_uint(mind[u]) >> 22;   // non-negative floats: monotone
  atomicAdd(&hist[b], 1);
}

__global__ void __launch_bounds__(1024) k_pick(int* hist, int* prm, int pass) {
  __shared__ int sc[1024];
  const int t = threadIdx.x;
  const int mine = hist[t];
  sc[t] = mine;
  __syncthreads();
  for (int off = 1; off < 1024; off <<= 1) {
    const int add = (t >= off) ? sc[t - off] : 0;
    __syncthreads();
    sc[t] += add;
    __syncthreads();
  }
  const int target = (pass == 0) ? NSEL : (NSEL - prm[1]);
  const int cum  = sc[t];
  const int prev = (t == 0) ? 0 : sc[t - 1];
  if (prev < target && cum >= target) {
    prm[pass * 2 + 0] = t;       // selected bin
    prm[pass * 2 + 1] = prev;    // count strictly below
  }
  hist[t] = 0;                   // reset for next pass / next call
}

__global__ void k_hist2(const float* __restrict__ mind, int* __restrict__ hist,
                        const int* __restrict__ prm) {
  const int u = blockIdx.x * 256 + threadIdx.x;
  const unsigned bits = __float_as_uint(mind[u]);
  if ((int)(bits >> 22) == prm[0]) atomicAdd(&hist[(bits >> 12) & 1023], 1);
}

// Deterministic compaction: strict-less first, then first `rem` ties by index.
__global__ void __launch_bounds__(1024) k_select(const float* __restrict__ mind,
                                                 const int* __restrict__ prm,
                                                 int* __restrict__ sel) {
  __shared__ int scs[1024], sct[1024];
  const int t = threadIdx.x;
  const unsigned thr12 = ((unsigned)prm[0] << 10) | (unsigned)prm[2];
  const int base = t * 64;
  int cs_ = 0, ct_ = 0;
  for (int i = 0; i < 64; ++i) {
    const unsigned b12 = __float_as_uint(mind[base + i]) >> 12;
    cs_ += (b12 < thr12);
    ct_ += (b12 == thr12);
  }
  scs[t] = cs_; sct[t] = ct_;
  __syncthreads();
  for (int off = 1; off < 1024; off <<= 1) {
    const int aS = (t >= off) ? scs[t - off] : 0;
    const int aT = (t >= off) ? sct[t - off] : 0;
    __syncthreads();
    scs[t] += aS; sct[t] += aT;
    __syncthreads();
  }
  const int S   = scs[1023];          // total strictly-below count
  const int rem = NSEL - S;           // ties to take (first by index)
  int ps = scs[t] - cs_;
  int pt = sct[t] - ct_;
  for (int i = 0; i < 64; ++i) {
    const int u = base + i;
    const unsigned b12 = __float_as_uint(mind[u]) >> 12;
    if (b12 < thr12) {
      sel[ps++] = u;
    } else if (b12 == thr12) {
      if (pt < rem) sel[S + pt] = u;
      pt++;
    }
  }
}

// ------------------------------------------------------------------
// Cross entropy: one block per row (1024 floats held in registers)
// ------------------------------------------------------------------
__global__ void k_ce(const float* __restrict__ outputs, const int* __restrict__ labels,
                     const int* __restrict__ mini, const int* __restrict__ sel,
                     float* __restrict__ nll) {
  __shared__ float red[256];
  __shared__ float xl;
  const int i = blockIdx.x;
  const int t = threadIdx.x;
  int rowi, lbl;
  if (i < NPOS) { rowi = i; lbl = labels[i]; }
  else { const int j = sel[i - NPOS]; rowi = NPOS + j; lbl = mini[j]; }

  const float4 v = ((const float4*)(outputs + (size_t)rowi * KDIM))[t];
  if ((lbl >> 2) == t) xl = ((const float*)&v)[lbl & 3];

  red[t] = fmaxf(fmaxf(v.x, v.y), fmaxf(v.z, v.w));
  __syncthreads();
  for (int s = 128; s > 0; s >>= 1) {
    if (t < s) red[t] = fmaxf(red[t], red[t + s]);
    __syncthreads();
  }
  const float rowmax = red[0];
  __syncthreads();
  red[t] = expf(v.x - rowmax) + expf(v.y - rowmax) + expf(v.z - rowmax) + expf(v.w - rowmax);
  __syncthreads();
  for (int s = 128; s > 0; s >>= 1) {
    if (t < s) red[t] += red[t + s];
    __syncthreads();
  }
  if (t == 0) nll[i] = logf(red[0]) + rowmax - xl;
}

__global__ void k_reduce(const float* __restrict__ nll, float* __restrict__ out) {
  __shared__ float red[256];
  const int t = threadIdx.x;
  float s = 0.f;
  for (int i = t; i < NTOT; i += 256) s += nll[i];  // fixed-order strided sums
  red[t] = s;
  __syncthreads();
  for (int k = 128; k > 0; k >>= 1) {
    if (t < k) red[t] += red[t + k];
    __syncthreads();
  }
  if (t == 0) out[0] = red[0] / (float)NTOT;
}

// ------------------------------------------------------------------
extern "C" void kernel_launch(void* const* d_in, const int* in_sizes, int n_in,
                              void* d_out, int out_size, void* d_ws, size_t ws_size,
                              hipStream_t stream) {
  const float* outputs = (const float*)d_in[0];   // [B, K] f32
  const int*   labels  = (const int*)d_in[1];     // [B] i32
  const float* glogit  = (const float*)d_in[2];   // [K, K] f32
  float* out = (float*)d_out;

  char* w = (char*)d_ws;
  _Float16* gl16 = (_Float16*)(w);                      // 2,097,152 B
  float* g2   = (float*)(w + 2097152);                  //     4,096 B
  float* mind = (float*)(w + 2101248);                  //   262,144 B
  int*   mini = (int*)  (w + 2363392);                  //   262,144 B
  int*   hist = (int*)  (w + 2625536);                  //     4,096 B
  int*   prm  = (int*)  (w + 2629632);                  //        64 B
  int*   sel  = (int*)  (w + 2629696);                  //    26,240 B
  float* nll  = (float*)(w + 2655936);                  //   288,356 B

  k_prep_gl<<<KDIM, 256, 0, stream>>>(glogit, gl16, g2);
  k_dist<<<NUNL / 64, 256, 0, stream>>>(outputs, gl16, g2, mind, mini);
  k_zero_hist<<<1, 1024, 0, stream>>>(hist, prm);
  k_hist1<<<NUNL / 256, 256, 0, stream>>>(mind, hist);
  k_pick<<<1, 1024, 0, stream>>>(hist, prm, 0);
  k_hist2<<<NUNL / 256, 256, 0, stream>>>(mind, hist, prm);
  k_pick<<<1, 1024, 0, stream>>>(hist, prm, 1);
  k_select<<<1, 1024, 0, stream>>>(mind, prm, sel);
  k_ce<<<NTOT, 256, 0, stream>>>(outputs, labels, mini, sel, nll);
  k_reduce<<<1, 256, 0, stream>>>(nll, out);
}
